// EmberBlock_77369540870297
// MI455X (gfx1250) — matrix operand order repaired and checked by reading, not compiled, not run on hardware
//
#include <hip/hip_runtime.h>
#include <hip/hip_bf16.h>
#include <math.h>

// Problem constants (reference: B=4, T=1024, C=768, H=12)
constexpr int EB_B  = 4;
constexpr int EB_T  = 1024;
constexpr int EB_C  = 768;
constexpr int EB_H  = 12;
constexpr int EB_HD = 64;
constexpr int EB_M  = EB_B * EB_T;   // 4096 rows

typedef _Float16 half_t;
typedef __attribute__((ext_vector_type(16))) _Float16 v16h;
typedef __attribute__((ext_vector_type(8)))  float    v8f;

__device__ __forceinline__ float sigmoidf_(float x) { return 1.0f / (1.0f + __expf(-x)); }
__device__ __forceinline__ float softplusf_(float x) { return (x > 20.0f) ? x : log1pf(__expf(x)); }
__device__ __forceinline__ float geluf_(float x) { return 0.5f * x * (1.0f + erff(x * 0.70710678118654752f)); }

__device__ __forceinline__ v8f wmma_f16(v16h a, v16h b, v8f c) {
    return __builtin_amdgcn_wmma_f32_16x16x32_f16(false, a, false, b, (short)0, c, false, false);
}

// ---- WMMA fragment loaders (layouts per CDNA5 ISA §7.12.2) ----

// A fragment: 16x32 f16 tile from row-major memory, leading dim lda.
// lane<16: M=lane, K pairs {0..7,16..23}; lane>=16: M=lane-16, K pairs {8..15,24..31}
__device__ __forceinline__ v16h load_a_f16(const half_t* A, int lda) {
    int lane = threadIdx.x & 31;
    int m  = lane & 15;
    int kb = (lane >> 4) << 3;
    v16h a;
#pragma unroll
    for (int r = 0; r < 8; ++r) {
        int k = ((r & 4) ? 16 : 0) + kb + ((r & 3) << 1);
        a[2 * r]     = A[(size_t)m * lda + k];
        a[2 * r + 1] = A[(size_t)m * lda + k + 1];
    }
    return a;
}

// Same A fragment but sourced from an f32 LDS tile (convert to f16 on load).
__device__ __forceinline__ v16h load_a_lds_f32(const float* P, int ld) {
    int lane = threadIdx.x & 31;
    int m  = lane & 15;
    int kb = (lane >> 4) << 3;
    v16h a;
#pragma unroll
    for (int r = 0; r < 8; ++r) {
        int k = ((r & 4) ? 16 : 0) + kb + ((r & 3) << 1);
        a[2 * r]     = (half_t)P[m * ld + k];
        a[2 * r + 1] = (half_t)P[m * ld + k + 1];
    }
    return a;
}

// B fragment: 32x16 f16 tile from row-major memory, leading dim ldb.
// lane L holds column N=L%16; K order matches the A fragment (lane<16: {0..7,16..23}, lane>=16: {8..15,24..31}).
__device__ __forceinline__ v16h load_b_f16(const half_t* Bp, int ldb) {
    int lane = threadIdx.x & 31;
    int n  = lane & 15;
    int kb = (lane >> 4) << 3;
    v16h b;
#pragma unroll
    for (int j = 0; j < 16; ++j) b[j] = Bp[(size_t)(((j & 8) ? 16 : 0) + kb + (j & 7)) * ldb + n];
    return b;
}

// ---------------- utility kernels ----------------

__global__ void f32_to_f16_kernel(const float* __restrict__ s, half_t* __restrict__ d, int n) {
    int i = blockIdx.x * blockDim.x + threadIdx.x;
    if (i < n) d[i] = (half_t)s[i];
}

__global__ void zero_f32_kernel(float* __restrict__ p, int n) {
    int i = blockIdx.x * blockDim.x + threadIdx.x;
    if (i < n) p[i] = 0.0f;
}

__global__ void finalize_ref_kernel(const float* __restrict__ refacc, float* __restrict__ out) {
    int i = blockIdx.x * blockDim.x + threadIdx.x;
    if (i < EB_B * EB_T) out[i] = refacc[i] * (1.0f / (float)(EB_H * EB_T));
}

// LayerNorm: one block (256 threads) per row of 768; writes f16.
__global__ __launch_bounds__(256) void ln_kernel(const float* __restrict__ x,
                                                 const float* __restrict__ w,
                                                 const float* __restrict__ b,
                                                 half_t* __restrict__ out) {
    __shared__ float red[256];
    int row = blockIdx.x;
    int tid = threadIdx.x;
    const float* xr = x + (size_t)row * EB_C;
    float s = 0.0f;
    for (int i = tid; i < EB_C; i += 256) s += xr[i];
    red[tid] = s; __syncthreads();
    for (int o = 128; o; o >>= 1) { if (tid < o) red[tid] += red[tid + o]; __syncthreads(); }
    float mean = red[0] / EB_C; __syncthreads();
    float vs = 0.0f;
    for (int i = tid; i < EB_C; i += 256) { float d = xr[i] - mean; vs += d * d; }
    red[tid] = vs; __syncthreads();
    for (int o = 128; o; o >>= 1) { if (tid < o) red[tid] += red[tid + o]; __syncthreads(); }
    float inv = rsqrtf(red[0] / EB_C + 1e-5f);
    for (int i = tid; i < EB_C; i += 256)
        out[(size_t)row * EB_C + i] = (half_t)((xr[i] - mean) * inv * w[i] + b[i]);
}

// ---------------- WMMA GEMM: one wave -> 32x64 output tile ----------------
// Two A fragments share the four B fragments -> 8 WMMAs per K=32 step,
// halving B-panel traffic per FLOP vs a 16-row tile.
// MODE 0: qkv   -> scatter q16 / kT16(transposed) / v16 (f16), +bias
// MODE 1: cproj -> outF = acc + bias + resid (fp32, x1)
// MODE 2: fc    -> outH = gelu(acc + bias) (f16)
// MODE 3: proj  -> outF = acc + bias + resid (fp32, final x)
template <int MODE>
__global__ __launch_bounds__(32) void gemm_wmma(const half_t* __restrict__ A,
                                                const half_t* __restrict__ W,
                                                const float* __restrict__ bias,
                                                int N, int K,
                                                const float* __restrict__ resid,
                                                float* __restrict__ outF,
                                                half_t* __restrict__ outH,
                                                half_t* __restrict__ q16,
                                                half_t* __restrict__ kT16,
                                                half_t* __restrict__ v16) {
    int col0 = blockIdx.x * 64;
    int row0 = blockIdx.y * 32;
    int lane = threadIdx.x & 31;

    v8f cacc[2][4] = {};
    for (int kk = 0; kk < K; kk += 32) {
        v16h a0 = load_a_f16(A + (size_t)row0 * K + kk, K);
        v16h a1 = load_a_f16(A + (size_t)(row0 + 16) * K + kk, K);
        const half_t* Wp = W + (size_t)kk * N + col0;
        v16h b0 = load_b_f16(Wp,      N);
        v16h b1 = load_b_f16(Wp + 16, N);
        v16h b2 = load_b_f16(Wp + 32, N);
        v16h b3 = load_b_f16(Wp + 48, N);
        cacc[0][0] = wmma_f16(a0, b0, cacc[0][0]);
        cacc[0][1] = wmma_f16(a0, b1, cacc[0][1]);
        cacc[0][2] = wmma_f16(a0, b2, cacc[0][2]);
        cacc[0][3] = wmma_f16(a0, b3, cacc[0][3]);
        cacc[1][0] = wmma_f16(a1, b0, cacc[1][0]);
        cacc[1][1] = wmma_f16(a1, b1, cacc[1][1]);
        cacc[1][2] = wmma_f16(a1, b2, cacc[1][2]);
        cacc[1][3] = wmma_f16(a1, b3, cacc[1][3]);
    }

    int nloc = lane & 15;
    int mb = (lane >> 4) << 3;
#pragma unroll
    for (int i = 0; i < 2; ++i) {
#pragma unroll
        for (int j = 0; j < 4; ++j) {
            int n = col0 + 16 * j + nloc;
#pragma unroll
            for (int r = 0; r < 8; ++r) {
                int m = row0 + 16 * i + mb + r;
                float v = cacc[i][j][r] + bias[n];
                if (MODE == 0) {
                    int b = m >> 10, t = m & (EB_T - 1);
                    int part = n / EB_C;
                    int cr = n - part * EB_C;
                    int h = cr >> 6, d = cr & 63;
                    size_t bh = (size_t)(b * EB_H + h);
                    half_t hv = (half_t)v;
                    if (part == 0)      q16[(bh * EB_T + t) * EB_HD + d]  = hv;
                    else if (part == 1) kT16[(bh * EB_HD + d) * EB_T + t] = hv;
                    else                v16[(bh * EB_T + t) * EB_HD + d]  = hv;
                } else if (MODE == 1) {
                    outF[(size_t)m * EB_C + n] = v + resid[(size_t)m * EB_C + n];
                } else if (MODE == 2) {
                    outH[(size_t)m * N + n] = (half_t)geluf_(v);
                } else {
                    outF[(size_t)m * EB_C + n] = v + resid[(size_t)m * EB_C + n];
                }
            }
        }
    }
}

// ---------------- attention pass 1: softmax stats + column load ----------------
// One wave per (qtile16, b*h). LDS: 16 x 1024 f32 probability row-block.
__global__ __launch_bounds__(32) void attn_pass1(const half_t* __restrict__ q16,
                                                 const half_t* __restrict__ kT16,
                                                 float* __restrict__ mrow,
                                                 float* __restrict__ lrow,
                                                 float* __restrict__ colsum) {
    extern __shared__ float P[];
    const int LD = EB_T;
    int qt = blockIdx.x;
    int bh = blockIdx.y;
    int lane = threadIdx.x & 31;
    int ktiles = qt + 1;
    int L = ktiles * 16;
    int q0 = qt * 16;
    int nloc = lane & 15, mb = (lane >> 4) << 3;

    const half_t* Qt = q16 + ((size_t)bh * EB_T + q0) * EB_HD;
    v16h a0 = load_a_f16(Qt, EB_HD);
    v16h a1 = load_a_f16(Qt + 32, EB_HD);
    const half_t* Kt = kT16 + (size_t)bh * EB_HD * EB_T;

    for (int kt = 0; kt < ktiles; ++kt) {
        v16h b0 = load_b_f16(Kt + kt * 16, EB_T);                     // d rows 0..31
        v16h b1 = load_b_f16(Kt + (size_t)32 * EB_T + kt * 16, EB_T); // d rows 32..63
        v8f c = {};
        c = wmma_f16(a0, b0, c);
        c = wmma_f16(a1, b1, c);
#pragma unroll
        for (int r = 0; r < 8; ++r) {
            int m = mb + r;
            int k = kt * 16 + nloc;
            float s = c[r] * 0.125f;               // 1/sqrt(64)
            if (k > q0 + m) s = -1e30f;            // causal mask
            P[m * LD + k] = s;
        }
    }
    __syncthreads();

    // row-wise softmax; save (max, sum)
    for (int m = 0; m < 16; ++m) {
        float mx = -1e30f;
        for (int k = lane; k < L; k += 32) mx = fmaxf(mx, P[m * LD + k]);
        for (int o = 16; o; o >>= 1) mx = fmaxf(mx, __shfl_xor(mx, o, 32));
        float s = 0.0f;
        for (int k = lane; k < L; k += 32) { float e = __expf(P[m * LD + k] - mx); P[m * LD + k] = e; s += e; }
        for (int o = 16; o; o >>= 1) s += __shfl_xor(s, o, 32);
        float inv = 1.0f / s;
        for (int k = lane; k < L; k += 32) P[m * LD + k] *= inv;
        if (lane == 0) {
            mrow[(size_t)bh * EB_T + q0 + m] = mx;
            lrow[(size_t)bh * EB_T + q0 + m] = s;
        }
    }
    // per-key column sums of probabilities (global accumulation over all q-tiles)
    for (int k = lane; k < L; k += 32) {
        float s = 0.0f;
#pragma unroll
        for (int m = 0; m < 16; ++m) s += P[m * LD + k];
        atomicAdd(&colsum[(size_t)bh * EB_T + k], s);
    }
}

// ---------------- attention pass 2: LIF modulation + renorm + PV ----------------
__global__ __launch_bounds__(32) void attn_pass2(const half_t* __restrict__ q16,
                                                 const half_t* __restrict__ kT16,
                                                 const half_t* __restrict__ v16,
                                                 const float* __restrict__ mrow,
                                                 const float* __restrict__ lrow,
                                                 const float* __restrict__ colsum,
                                                 const float* __restrict__ prev,
                                                 const float* __restrict__ thr_p,
                                                 const float* __restrict__ leak_p,
                                                 const float* __restrict__ steep_p,
                                                 const float* __restrict__ refs_p,
                                                 const float* __restrict__ cross_p,
                                                 float* __restrict__ refacc,
                                                 half_t* __restrict__ y16) {
    extern __shared__ float P[];
    const int LD = EB_T;
    int qt = blockIdx.x;
    int bh = blockIdx.y;
    int b = bh / EB_H, h = bh % EB_H;
    int lane = threadIdx.x & 31;
    int ktiles = qt + 1;
    int L = ktiles * 16;
    int q0 = qt * 16;
    int nloc = lane & 15, mb = (lane >> 4) << 3;

    // recompute probabilities from stored softmax stats
    const half_t* Qt = q16 + ((size_t)bh * EB_T + q0) * EB_HD;
    v16h a0 = load_a_f16(Qt, EB_HD);
    v16h a1 = load_a_f16(Qt + 32, EB_HD);
    const half_t* Kt = kT16 + (size_t)bh * EB_HD * EB_T;

    float mxv[8], ilv[8];
#pragma unroll
    for (int r = 0; r < 8; ++r) {
        mxv[r] = mrow[(size_t)bh * EB_T + q0 + mb + r];
        ilv[r] = 1.0f / lrow[(size_t)bh * EB_T + q0 + mb + r];
    }

    for (int kt = 0; kt < ktiles; ++kt) {
        v16h b0 = load_b_f16(Kt + kt * 16, EB_T);
        v16h b1 = load_b_f16(Kt + (size_t)32 * EB_T + kt * 16, EB_T);
        v8f c = {};
        c = wmma_f16(a0, b0, c);
        c = wmma_f16(a1, b1, c);
#pragma unroll
        for (int r = 0; r < 8; ++r) {
            int m = mb + r;
            int k = kt * 16 + nloc;
            float p = (k > q0 + m) ? 0.0f : __expf(c[r] * 0.125f - mxv[r]) * ilv[r];
            P[m * LD + k] = p;
        }
    }
    __syncthreads();

    // per-head LIF parameters
    float thr = fabsf(thr_p[h]) * 0.1f;
    float lk  = sigmoidf_(leak_p[h]);
    float stp = softplusf_(steep_p[h]);
    float rs  = softplusf_(refs_p[h]);
    float cw  = sigmoidf_(cross_p[h]);

    // modulation + row renormalization (in LDS)
    for (int m = 0; m < 16; ++m) {
        float s = 0.0f;
        for (int k = lane; k < L; k += 32) {
            float p = P[m * LD + k];
            float eff = thr + rs * (colsum[(size_t)bh * EB_T + k] * (1.0f / EB_T))
                            + cw * prev[(size_t)b * EB_T + k];
            float fm = sigmoidf_(stp * (p - eff));
            float mod = p * (fm + lk * (1.0f - fm));
            P[m * LD + k] = mod;
            s += mod;
        }
        for (int o = 16; o; o >>= 1) s += __shfl_xor(s, o, 32);
        float inv = 1.0f / (s + 1e-8f);
        for (int k = lane; k < L; k += 32) P[m * LD + k] *= inv;
    }

    // new_refractory accumulation: sum over (h, q) per key column
    for (int k = lane; k < L; k += 32) {
        float s = 0.0f;
#pragma unroll
        for (int m = 0; m < 16; ++m) s += P[m * LD + k];
        atomicAdd(&refacc[(size_t)b * EB_T + k], s);
    }

    // zero-pad one extra k-tile if odd tile count (K step of PV is 32)
    if (ktiles & 1) {
        for (int i = lane; i < 256; i += 32) P[(i >> 4) * LD + L + (i & 15)] = 0.0f;
    }
    __syncthreads();

    // PV: y(16x64) = modulated(16xL) @ V(Lx64) via WMMA, A sourced from LDS
    v8f cy[4] = {};
    const half_t* Vt = v16 + (size_t)bh * EB_T * EB_HD;
    int kpad = ((ktiles + 1) & ~1) * 16;
    for (int kk = 0; kk < kpad; kk += 32) {
        v16h a = load_a_lds_f32(P + kk, LD);
        const half_t* Vp = Vt + (size_t)kk * EB_HD;
        v16h b0 = load_b_f16(Vp,      EB_HD);
        v16h b1 = load_b_f16(Vp + 16, EB_HD);
        v16h b2 = load_b_f16(Vp + 32, EB_HD);
        v16h b3 = load_b_f16(Vp + 48, EB_HD);
        cy[0] = wmma_f16(a, b0, cy[0]);
        cy[1] = wmma_f16(a, b1, cy[1]);
        cy[2] = wmma_f16(a, b2, cy[2]);
        cy[3] = wmma_f16(a, b3, cy[3]);
    }
#pragma unroll
    for (int j = 0; j < 4; ++j) {
#pragma unroll
        for (int r = 0; r < 8; ++r) {
            int q = q0 + mb + r;
            int d = 16 * j + nloc;
            y16[((size_t)b * EB_T + q) * EB_C + h * EB_HD + d] = (half_t)cy[j][r];
        }
    }
}

// ---------------- host-side launch ----------------

extern "C" void kernel_launch(void* const* d_in, const int* in_sizes, int n_in,
                              void* d_out, int out_size, void* d_ws, size_t ws_size,
                              hipStream_t stream) {
    (void)in_sizes; (void)n_in; (void)out_size; (void)ws_size;
    const float* x        = (const float*)d_in[0];
    const float* refstate = (const float*)d_in[1];
    const float* ln1w     = (const float*)d_in[2];
    const float* ln1b     = (const float*)d_in[3];
    const float* attnW    = (const float*)d_in[4];
    const float* attnB    = (const float*)d_in[5];
    const float* cprojW   = (const float*)d_in[6];
    const float* cprojB   = (const float*)d_in[7];
    const float* thr      = (const float*)d_in[8];
    const float* leak     = (const float*)d_in[9];
    const float* steep    = (const float*)d_in[10];
    const float* refs     = (const float*)d_in[11];
    const float* cross    = (const float*)d_in[12];
    const float* ln2w     = (const float*)d_in[13];
    const float* ln2b     = (const float*)d_in[14];
    const float* fcW      = (const float*)d_in[15];
    const float* fcB      = (const float*)d_in[16];
    const float* prW      = (const float*)d_in[17];
    const float* prB      = (const float*)d_in[18];

    char* ws = (char*)d_ws;
    size_t off = 0;
    auto carve = [&](size_t bytes) -> void* {
        void* p = ws + off;
        off += (bytes + 255) & ~(size_t)255;
        return p;
    };
    half_t* wAttn16  = (half_t*)carve((size_t)EB_C * 3 * EB_C * sizeof(half_t));
    half_t* wCproj16 = (half_t*)carve((size_t)EB_C * EB_C * sizeof(half_t));
    half_t* wFc16    = (half_t*)carve((size_t)EB_C * 4 * EB_C * sizeof(half_t));
    half_t* wPr16    = (half_t*)carve((size_t)4 * EB_C * EB_C * sizeof(half_t));
    half_t* h16      = (half_t*)carve((size_t)EB_M * EB_C * sizeof(half_t));
    half_t* q16      = (half_t*)carve((size_t)EB_M * EB_C * sizeof(half_t));
    half_t* kT16     = (half_t*)carve((size_t)EB_M * EB_C * sizeof(half_t));
    half_t* v16      = (half_t*)carve((size_t)EB_M * EB_C * sizeof(half_t));
    half_t* y16      = (half_t*)carve((size_t)EB_M * EB_C * sizeof(half_t));
    half_t* h2_16    = (half_t*)carve((size_t)EB_M * EB_C * sizeof(half_t));
    half_t* a16      = (half_t*)carve((size_t)EB_M * 4 * EB_C * sizeof(half_t));
    float*  x1       = (float*)carve((size_t)EB_M * EB_C * sizeof(float));
    float*  mrow     = (float*)carve((size_t)EB_B * EB_H * EB_T * sizeof(float));
    float*  lrow     = (float*)carve((size_t)EB_B * EB_H * EB_T * sizeof(float));
    float*  colsum   = (float*)carve((size_t)EB_B * EB_H * EB_T * sizeof(float));
    float*  refacc   = (float*)carve((size_t)EB_B * EB_T * sizeof(float));

    float* out_x   = (float*)d_out;
    float* out_ref = (float*)d_out + (size_t)EB_M * EB_C;

    auto cvt = [&](const float* s, half_t* d, int n) {
        f32_to_f16_kernel<<<(n + 255) / 256, 256, 0, stream>>>(s, d, n);
    };
    // weight conversions (stateless, every call)
    cvt(attnW, wAttn16, EB_C * 3 * EB_C);
    cvt(cprojW, wCproj16, EB_C * EB_C);
    cvt(fcW, wFc16, EB_C * 4 * EB_C);
    cvt(prW, wPr16, 4 * EB_C * EB_C);

    // zero accumulators
    zero_f32_kernel<<<(EB_B * EB_H * EB_T + 255) / 256, 256, 0, stream>>>(colsum, EB_B * EB_H * EB_T);
    zero_f32_kernel<<<(EB_B * EB_T + 255) / 256, 256, 0, stream>>>(refacc, EB_B * EB_T);

    // LN1
    ln_kernel<<<EB_M, 256, 0, stream>>>(x, ln1w, ln1b, h16);

    // qkv GEMM + scatter (N=2304)
    gemm_wmma<0><<<dim3(3 * EB_C / 64, EB_M / 32), 32, 0, stream>>>(
        h16, wAttn16, attnB, 3 * EB_C, EB_C, nullptr, nullptr, nullptr, q16, kT16, v16);

    // attention
    const size_t lds = (size_t)16 * EB_T * sizeof(float);  // 64 KB
    attn_pass1<<<dim3(EB_T / 16, EB_B * EB_H), 32, lds, stream>>>(q16, kT16, mrow, lrow, colsum);
    attn_pass2<<<dim3(EB_T / 16, EB_B * EB_H), 32, lds, stream>>>(
        q16, kT16, v16, mrow, lrow, colsum, refstate,
        thr, leak, steep, refs, cross, refacc, y16);

    // c_proj + residual -> x1 (fp32)
    gemm_wmma<1><<<dim3(EB_C / 64, EB_M / 32), 32, 0, stream>>>(
        y16, wCproj16, cprojB, EB_C, EB_C, x, x1, nullptr, nullptr, nullptr, nullptr);

    // LN2
    ln_kernel<<<EB_M, 256, 0, stream>>>(x1, ln2w, ln2b, h2_16);

    // fc + exact GELU -> a16 (N=3072)
    gemm_wmma<2><<<dim3(4 * EB_C / 64, EB_M / 32), 32, 0, stream>>>(
        h2_16, wFc16, fcB, 4 * EB_C, EB_C, nullptr, nullptr, a16, nullptr, nullptr, nullptr);

    // proj + residual -> final x (fp32 into d_out)
    gemm_wmma<3><<<dim3(EB_C / 64, EB_M / 32), 32, 0, stream>>>(
        a16, wPr16, prB, EB_C, 4 * EB_C, x1, out_x, nullptr, nullptr, nullptr, nullptr);

    // new_refractory = refacc / (H*T)
    finalize_ref_kernel<<<(EB_B * EB_T + 255) / 256, 256, 0, stream>>>(refacc, out_ref);
}
